// Transducer_60198261621429
// MI455X (gfx1250) — compile-verified, matrix operand order checked by reading, not run on hardware
//
#include <hip/hip_runtime.h>
#include <hip/hip_bf16.h>
#include <hip/hip_fp16.h>

// ---------------- CDNA5 WMMA fragment helpers (wave32, 16x16x32 f16) -------
typedef __attribute__((ext_vector_type(16))) _Float16 v16h;
typedef __attribute__((ext_vector_type(8)))  _Float16 v8h;
typedef __attribute__((ext_vector_type(8)))  float    v8f;
typedef __attribute__((ext_vector_type(4)))  unsigned int v4u;
typedef __attribute__((ext_vector_type(8)))  int      v8i;
typedef __attribute__((ext_vector_type(4)))  int      v4i;

#define WMMA_F16(a, b, c) \
  __builtin_amdgcn_wmma_f32_16x16x32_f16(false, (a), false, (b), (short)0, (c), false, false)

// CDNA5 16-bit operand layout: per lane, K-halves live in two contiguous
// 16-byte chunks: lanes 0-15 hold K[k0+0..7] and K[k0+16..23]; lanes 16-31
// hold K[k0+8..15] and K[k0+24..31].  => two b128 loads, zero guards.
__device__ __forceinline__ v16h frag_f16(const _Float16* __restrict__ src, int ld,
                                         int r0, int k0, int lane) {
  int r  = r0 + (lane & 15);
  int hi = lane >> 4;
  const _Float16* p = src + (long)r * ld + k0 + hi * 8;
  v8h lo = *(const v8h*)(p);
  v8h up = *(const v8h*)(p + 16);
  return __builtin_shufflevector(lo, up, 0, 1, 2, 3, 4, 5, 6, 7,
                                         8, 9, 10, 11, 12, 13, 14, 15);
}

__device__ __forceinline__ float sigf(float x) { return 1.f / (1.f + expf(-x)); }

// ---------------- TDM gather: 16 rows (256 f16 each) -> LDS, 1024B pitch ----
// D# per CDNA5 ISA ch.8: group0 = {flags, lds_addr, global_addr, type=2};
// group1 = {data_size=2B | pad_enable | pad_interval=128dw | pad_amount=128dw,
//           tensor_dim0=256, tensor_dim1=rows, tile_dim0=256, tile_dim1=16,
//           tensor_dim0_stride=256}; groups 2/3 = 16x 16-bit row indices.
__device__ __forceinline__ void tdm_gather16_rows(const _Float16* tensor, int tensor_rows,
                                                  unsigned lds_byte_addr,
                                                  const unsigned* pk /*8 packed dwords*/) {
  unsigned long long ga = (unsigned long long)(size_t)tensor;
  v4u g0 = { 1u | (1u << 31),                                   // count=1, gather, 16b idx
             lds_byte_addr,
             (unsigned)(ga & 0xFFFFFFFFull),
             (unsigned)((ga >> 32) & 0x1FFFFFFull) | (2u << 30) }; // type=2 (image)
  v8i g1 = { (int)((1u << 16) | (1u << 20) | (6u << 22) | (127u << 25)),
             (int)(256u << 16),                                  // tensor_dim0 = 256
             (int)(((unsigned)tensor_rows & 0xFFFFu) << 16),     // tensor_dim1
             (int)(256u << 16),                                  // tile_dim0 = 256
             16,                                                 // tile_dim1 = #indices
             256, 0, 0 };                                        // dim0_stride = 256
  v4i g2 = { (int)pk[0], (int)pk[1], (int)pk[2], (int)pk[3] };
  v4i g3 = { (int)pk[4], (int)pk[5], (int)pk[6], (int)pk[7] };
  v8i gz = { 0, 0, 0, 0, 0, 0, 0, 0 };
  __builtin_amdgcn_tensor_load_to_lds(g0, g1, g2, g3, gz, 0);
}

// ---------------- pad + convert f32[M,K] -> f16[Mp,Kp] (zero fill) ----------
__global__ void pad_cvt_f16(const float* __restrict__ src, _Float16* __restrict__ dst,
                            int M, int K, int Mp, int Kp) {
  int i = blockIdx.x * blockDim.x + threadIdx.x;
  if (i >= Mp * Kp) return;
  int r = i / Kp, c = i - r * Kp;
  float v = (r < M && c < K) ? src[(long)r * K + c] : 0.f;
  dst[i] = (_Float16)v;
}

// ---------------- WMMA GEMM: C[M,N] = A[Mp,Kp](f16) * W[N,Kp](f16)^T + bias -
__global__ __launch_bounds__(128)
void wmma_gemm(const _Float16* __restrict__ A, const _Float16* __restrict__ W,
               const float* __restrict__ b1, const float* __restrict__ b2,
               float* __restrict__ Cf, _Float16* __restrict__ Ch,
               int M, int N, int Kp) {
  int lane = threadIdx.x & 31;
  int wave = threadIdx.x >> 5;
  int Mt = (M + 15) >> 4, Nt = N >> 4;
  int tile = blockIdx.x * 4 + wave;
  if (tile >= Mt * Nt) return;
  int mt = tile / Nt, nt = tile % Nt;
  int m0 = mt * 16, n0 = nt * 16;
  int col = n0 + (lane & 15);
  int hi = lane >> 4;

  float bias = (b1 ? b1[col] : 0.f) + (b2 ? b2[col] : 0.f);
  v8f acc;
#pragma unroll
  for (int j = 0; j < 8; ++j) acc[j] = bias;

  for (int kk = 0; kk < (Kp >> 5); ++kk) {
    v16h a = frag_f16(A, Kp, m0, kk * 32, lane);
    v16h w = frag_f16(W, Kp, n0, kk * 32, lane);
    acc = WMMA_F16(a, w, acc);
  }
#pragma unroll
  for (int j = 0; j < 8; ++j) {
    int row = m0 + j + 8 * hi;
    if (row < M) {
      if (Cf) Cf[(long)row * N + col] = acc[j];
      else    Ch[(long)row * N + col] = (_Float16)acc[j];
    }
  }
}

// ---------------- persistent LSTM recurrence scan (B=8, H=256) --------------
__global__ __launch_bounds__(512)
void lstm_scan(const float* __restrict__ xp, const _Float16* __restrict__ Whh16,
               _Float16* __restrict__ h_out, int steps) {
  __shared__ __attribute__((aligned(16))) _Float16 h16[16 * 256]; // rows 8..15 zero
  __shared__ float cst[8 * 256];
  __shared__ float gbuf[8 * 1024];
  int tid = threadIdx.x, lane = tid & 31, wave = tid >> 5;

  for (int i = tid; i < 16 * 256; i += 512) h16[i] = (_Float16)0.f;
  for (int i = tid; i < 8 * 256; i += 512) cst[i] = 0.f;
  __syncthreads();

  for (int t = 0; t < steps; ++t) {
#pragma unroll
    for (int q = 0; q < 4; ++q) {       // 64 N-tiles; 16 waves x 4
      int n0 = (wave * 4 + q) * 16;
      int col = n0 + (lane & 15);
      int hi = lane >> 4;
      v8f acc;
#pragma unroll
      for (int j = 0; j < 8; ++j) acc[j] = 0.f;
#pragma unroll
      for (int kk = 0; kk < 8; ++kk) {
        v16h a = frag_f16(h16, 256, 0, kk * 32, lane);
        v16h w = frag_f16(Whh16, 256, n0, kk * 32, lane);
        acc = WMMA_F16(a, w, acc);
      }
      if (hi == 0) {                    // rows 0..7 are the real batch
#pragma unroll
        for (int j = 0; j < 8; ++j) gbuf[j * 1024 + col] = acc[j];
      }
    }
    __syncthreads();
    for (int i = tid; i < 8 * 256; i += 512) {
      int b = i >> 8, j = i & 255;
      long xrow = ((long)b * steps + t) * 1024;
      float gi = gbuf[b * 1024 + j]       + xp[xrow + j];
      float gf = gbuf[b * 1024 + 256 + j] + xp[xrow + 256 + j];
      float gg = gbuf[b * 1024 + 512 + j] + xp[xrow + 512 + j];
      float go = gbuf[b * 1024 + 768 + j] + xp[xrow + 768 + j];
      float c = sigf(gf) * cst[i] + sigf(gi) * tanhf(gg);
      float h = sigf(go) * tanhf(c);
      cst[i] = c;
      _Float16 h_16 = (_Float16)h;
      h16[b * 256 + j] = h_16;
      h_out[((long)b * steps + t) * 256 + j] = h_16;
    }
    __syncthreads();
  }
}

// ---------------- decoder embedding lookup (f16, padded to 528 rows) --------
__global__ void embed_lookup(const float* __restrict__ embed,
                             const int* __restrict__ ys,
                             _Float16* __restrict__ dec_in) {
  int i = blockIdx.x * blockDim.x + threadIdx.x;
  if (i >= 528 * 64) return;
  int e = i & 63, cell = i >> 6;
  float v = 0.f;
  if (cell < 520) {
    int u = cell % 65, b = cell / 65;
    int tok = (u == 0) ? 0 : ys[b * 64 + (u - 1)];
    v = embed[tok * 64 + e];
  }
  dec_in[i] = (_Float16)v;
}

// ---------------- fused joint network + log-softmax gather ------------------
// feat tile staged by TDM gather-mode loads (f rows -> low half, g rows ->
// high half of each 1024B LDS row via pad_amount).
__global__ __launch_bounds__(128)
void joint_kernel(const _Float16* __restrict__ f_enc, const _Float16* __restrict__ g_dec,
                  const _Float16* __restrict__ fc1w, const float* __restrict__ fc1b,
                  const _Float16* __restrict__ fc2w, const float* __restrict__ fc2b,
                  const int* __restrict__ ys,
                  float* __restrict__ lp_blank, float* __restrict__ lp_y) {
  __shared__ __attribute__((aligned(16))) _Float16 feat[16 * 512];
  __shared__ __attribute__((aligned(16))) _Float16 hid[16 * 256];
  __shared__ float logits[16 * 128];
  __shared__ int bb[16], tt[16], uu[16];

  int tid = threadIdx.x, lane = tid & 31, wave = tid >> 5;
  int m0 = blockIdx.x * 16;

  if (tid < 16) {
    int rr = m0 + tid;
    int u = rr % 65, bt = rr / 65;
    bb[tid] = bt >> 8; tt[tid] = bt & 255; uu[tid] = u;
  }
  if (tid == 0) {
    // pack 16x 16-bit row indices for each gather (non-decreasing f indices)
    unsigned fpk[8], gpk[8];
#pragma unroll
    for (int p = 0; p < 8; ++p) {
      int r0 = m0 + 2 * p, r1 = r0 + 1;
      unsigned f0 = (unsigned)(r0 / 65), f1 = (unsigned)(r1 / 65);
      unsigned g0 = (f0 >> 8) * 65u + (unsigned)(r0 % 65);
      unsigned g1 = (f1 >> 8) * 65u + (unsigned)(r1 % 65);
      fpk[p] = f0 | (f1 << 16);
      gpk[p] = g0 | (g1 << 16);
    }
    unsigned feat_lds = (unsigned)(size_t)(void*)feat;
    tdm_gather16_rows(f_enc, 2048, feat_lds, fpk);          // -> feat[r][0:256]
    tdm_gather16_rows(g_dec, 520, feat_lds + 512, gpk);     // -> feat[r][256:512]
    __builtin_amdgcn_s_wait_tensorcnt(0);
  }
  __syncthreads();

  // fc1: 16x512 -> 16x256 (16 N-tiles, 4/wave, K=512)
#pragma unroll
  for (int q = 0; q < 4; ++q) {
    int n0 = (wave + q * 4) * 16;
    int col = n0 + (lane & 15);
    int hi = lane >> 4;
    float bv = fc1b[col];
    v8f acc;
#pragma unroll
    for (int j = 0; j < 8; ++j) acc[j] = bv;
#pragma unroll
    for (int kk = 0; kk < 16; ++kk) {
      v16h a = frag_f16(feat, 512, 0, kk * 32, lane);
      v16h w = frag_f16(fc1w, 512, n0, kk * 32, lane);
      acc = WMMA_F16(a, w, acc);
    }
#pragma unroll
    for (int j = 0; j < 8; ++j)
      hid[(j + 8 * hi) * 256 + col] = (_Float16)tanhf(acc[j]);
  }
  __syncthreads();

  // fc2: 16x256 -> 16x128 (8 N-tiles, 2/wave, K=256)
#pragma unroll
  for (int q = 0; q < 2; ++q) {
    int n0 = (wave + q * 4) * 16;
    int col = n0 + (lane & 15);
    int hi = lane >> 4;
    float bv = fc2b[col];
    v8f acc;
#pragma unroll
    for (int j = 0; j < 8; ++j) acc[j] = bv;
#pragma unroll
    for (int kk = 0; kk < 8; ++kk) {
      v16h a = frag_f16(hid, 256, 0, kk * 32, lane);
      v16h w = frag_f16(fc2w, 256, n0, kk * 32, lane);
      acc = WMMA_F16(a, w, acc);
    }
#pragma unroll
    for (int j = 0; j < 8; ++j)
      logits[(j + 8 * hi) * 128 + col] = acc[j];
  }
  __syncthreads();

  if (tid < 16) {
    int row = tid;
    float mx = -1e30f;
    for (int j = 0; j < 128; ++j) mx = fmaxf(mx, logits[row * 128 + j]);
    float s = 0.f;
    for (int j = 0; j < 128; ++j) s += expf(logits[row * 128 + j] - mx);
    float lse = mx + logf(s);
    int b = bb[row], t = tt[row], u = uu[row];
    lp_blank[((long)b * 256 + t) * 65 + u] = logits[row * 128 + 0] - lse;
    if (u < 64) {
      int lab = ys[b * 64 + u];
      lp_y[((long)b * 256 + t) * 64 + u] = logits[row * 128 + lab] - lse;
    }
  }
}

// ---------------- RNN-T alpha DP: anti-diagonal wavefront per batch ---------
__global__ __launch_bounds__(96)
void rnnt_alpha(const float* __restrict__ lp_blank, const float* __restrict__ lp_y,
                float* __restrict__ alpha) {
  const int T = 256, U = 64, U1 = 65;
  int b = blockIdx.x;
  int u = threadIdx.x;
  __shared__ float dprev[65];

  for (int d = 0; d < T + U; ++d) {
    int t = d - u;
    bool active = (u <= U) && (t >= 0) && (t < T);
    float val = 0.f;
    if (active) {
      if (t == 0 && u == 0) {
        val = 0.f;
      } else if (t == 0) {
        val = dprev[u - 1] + lp_y[((long)b * T + 0) * U + (u - 1)];
      } else if (u == 0) {
        val = dprev[0] + lp_blank[((long)b * T + (t - 1)) * U1 + 0];
      } else {
        float x = dprev[u] + lp_blank[((long)b * T + (t - 1)) * U1 + u];
        float y = dprev[u - 1] + lp_y[((long)b * T + t) * U + (u - 1)];
        float mx = fmaxf(x, y);
        val = mx + logf(expf(x - mx) + expf(y - mx));
      }
      alpha[((long)b * T + t) * U1 + u] = val;
    }
    __syncthreads();
    if (active) dprev[u] = val;
    __syncthreads();
  }
}

// ---------------- final loss reduction --------------------------------------
__global__ void final_loss(const float* __restrict__ alpha,
                           const float* __restrict__ lp_blank,
                           const int* __restrict__ xlen, const int* __restrict__ ylen,
                           float* __restrict__ out) {
  if (threadIdx.x == 0 && blockIdx.x == 0) {
    float s = 0.f;
    for (int b = 0; b < 8; ++b) {
      int t = xlen[b] - 1, u = ylen[b];
      long idx = ((long)b * 256 + t) * 65 + u;
      s += alpha[idx] + lp_blank[idx];
    }
    out[0] = -s / 8.f;
  }
}

// ---------------- host orchestration ----------------------------------------
extern "C" void kernel_launch(void* const* d_in, const int* in_sizes, int n_in,
                              void* d_out, int out_size, void* d_ws, size_t ws_size,
                              hipStream_t stream) {
  const float* xs       = (const float*)d_in[0];
  const int*   ys       = (const int*)d_in[1];
  const int*   xlen     = (const int*)d_in[2];
  const int*   ylen     = (const int*)d_in[3];
  const float* enc_Wih0 = (const float*)d_in[4];
  const float* enc_Whh0 = (const float*)d_in[5];
  const float* enc_bih0 = (const float*)d_in[6];
  const float* enc_bhh0 = (const float*)d_in[7];
  const float* enc_Wih1 = (const float*)d_in[8];
  const float* enc_Whh1 = (const float*)d_in[9];
  const float* enc_bih1 = (const float*)d_in[10];
  const float* enc_bhh1 = (const float*)d_in[11];
  const float* enc_lW   = (const float*)d_in[12];
  const float* enc_lb   = (const float*)d_in[13];
  const float* embed    = (const float*)d_in[14];
  const float* dec_Wih  = (const float*)d_in[15];
  const float* dec_Whh  = (const float*)d_in[16];
  const float* dec_bih  = (const float*)d_in[17];
  const float* dec_bhh  = (const float*)d_in[18];
  const float* fc1_W    = (const float*)d_in[19];
  const float* fc1_b    = (const float*)d_in[20];
  const float* fc2_W    = (const float*)d_in[21];
  const float* fc2_b    = (const float*)d_in[22];

  char* ws = (char*)d_ws;
  auto alloc = [&](size_t bytes) -> void* {
    void* p = (void*)ws;
    ws += (bytes + 255) & ~(size_t)255;
    return p;
  };
  _Float16* xs16     = (_Float16*)alloc((size_t)2048 * 96 * 2);
  _Float16* wih0_16  = (_Float16*)alloc((size_t)1024 * 96 * 2);
  _Float16* wih1_16  = (_Float16*)alloc((size_t)1024 * 256 * 2);
  _Float16* lw16     = (_Float16*)alloc((size_t)256 * 256 * 2);
  _Float16* dwih16   = (_Float16*)alloc((size_t)1024 * 64 * 2);
  _Float16* whh0_16  = (_Float16*)alloc((size_t)1024 * 256 * 2);
  _Float16* whh1_16  = (_Float16*)alloc((size_t)1024 * 256 * 2);
  _Float16* dwhh_16  = (_Float16*)alloc((size_t)1024 * 256 * 2);
  _Float16* fc1w16   = (_Float16*)alloc((size_t)256 * 512 * 2);
  _Float16* fc2w16   = (_Float16*)alloc((size_t)128 * 256 * 2);
  float*    xp0      = (float*)alloc((size_t)2048 * 1024 * 4);
  float*    xp1      = (float*)alloc((size_t)2048 * 1024 * 4);
  float*    xp_dec   = (float*)alloc((size_t)528 * 1024 * 4);
  _Float16* h0_16    = (_Float16*)alloc((size_t)2048 * 256 * 2);
  _Float16* h1_16    = (_Float16*)alloc((size_t)2048 * 256 * 2);
  _Float16* f_enc16  = (_Float16*)alloc((size_t)2048 * 256 * 2);
  _Float16* dec_in16 = (_Float16*)alloc((size_t)528 * 64 * 2);
  _Float16* g_dec16  = (_Float16*)alloc((size_t)520 * 256 * 2);
  float*    lpb      = (float*)alloc((size_t)8 * 256 * 65 * 4);
  float*    lpy      = (float*)alloc((size_t)8 * 256 * 64 * 4);
  float*    alpha    = (float*)alloc((size_t)8 * 256 * 65 * 4);

  auto cvt = [&](const float* s, _Float16* d, int M, int K, int Mp, int Kp) {
    int n = Mp * Kp;
    pad_cvt_f16<<<(n + 255) / 256, 256, 0, stream>>>(s, d, M, K, Mp, Kp);
  };
  cvt(xs,       xs16,    2048, 80, 2048, 96);
  cvt(enc_Wih0, wih0_16, 1024, 80, 1024, 96);
  cvt(enc_Wih1, wih1_16, 1024, 256, 1024, 256);
  cvt(enc_lW,   lw16,    256, 256, 256, 256);
  cvt(dec_Wih,  dwih16,  1024, 64, 1024, 64);
  cvt(enc_Whh0, whh0_16, 1024, 256, 1024, 256);
  cvt(enc_Whh1, whh1_16, 1024, 256, 1024, 256);
  cvt(dec_Whh,  dwhh_16, 1024, 256, 1024, 256);
  cvt(fc1_W,    fc1w16,  256, 512, 256, 512);
  cvt(fc2_W,    fc2w16,  128, 256, 128, 256);

  auto gemm = [&](const _Float16* A, const _Float16* W, const float* b1,
                  const float* b2, float* Cf, _Float16* Ch, int M, int N, int Kp) {
    int tiles = ((M + 15) / 16) * (N / 16);
    wmma_gemm<<<(tiles + 3) / 4, 128, 0, stream>>>(A, W, b1, b2, Cf, Ch, M, N, Kp);
  };

  // encoder
  gemm(xs16, wih0_16, enc_bih0, enc_bhh0, xp0, nullptr, 2048, 1024, 96);
  lstm_scan<<<1, 512, 0, stream>>>(xp0, whh0_16, h0_16, 256);
  gemm(h0_16, wih1_16, enc_bih1, enc_bhh1, xp1, nullptr, 2048, 1024, 256);
  lstm_scan<<<1, 512, 0, stream>>>(xp1, whh1_16, h1_16, 256);
  gemm(h1_16, lw16, enc_lb, nullptr, nullptr, f_enc16, 2048, 256, 256);

  // decoder
  embed_lookup<<<(528 * 64 + 255) / 256, 256, 0, stream>>>(embed, ys, dec_in16);
  gemm(dec_in16, dwih16, dec_bih, dec_bhh, xp_dec, nullptr, 520, 1024, 64);
  lstm_scan<<<1, 512, 0, stream>>>(xp_dec, dwhh_16, g_dec16, 65);

  // fused joint + gathered log-softmax (133120 cells / 16 per block)
  joint_kernel<<<8320, 128, 0, stream>>>(f_enc16, g_dec16, fc1w16, fc1_b,
                                         fc2w16, fc2_b, ys, lpb, lpy);

  // RNN-T DP + final reduction
  rnnt_alpha<<<8, 96, 0, stream>>>(lpb, lpy, alpha);
  final_loss<<<1, 32, 0, stream>>>(alpha, lpb, xlen, ylen, (float*)d_out);
}
